// PANEncoderLayer_cross_32658931319253
// MI455X (gfx1250) — matrix operand order repaired and checked by reading, not compiled, not run on hardware
//
#include <hip/hip_runtime.h>

// ---------------- problem constants ----------------
static constexpr int B  = 4;
static constexpr int C  = 256;
static constexpr int H  = 128;
static constexpr int W  = 128;
static constexpr int HP = 32;          // pooled H
static constexpr int WP = 32;          // pooled W
static constexpr int L  = HP * WP;     // 1024 tokens
static constexpr int S  = L;
static constexpr int NH = 8;
static constexpr int D  = C / NH;      // 32
static constexpr float EPS = 1e-5f;
static constexpr int C2 = 2 * C;       // 512 (MLP hidden)

typedef __attribute__((ext_vector_type(16))) _Float16 v16h;
typedef __attribute__((ext_vector_type(8)))  float    v8f;

#define DEVINL __device__ __forceinline__

// ---------------- WMMA fragment helpers (gfx1250 wave32 layouts) -------------
// A-matrix 16x32 f16 (ISA 7.12.2): lanes 0-15 rows M=0..15 with K={0..7,16..23},
// lanes 16-31 same rows with K={8..15,24..31}. Two contiguous 16B runs per lane.
DEVINL v16h frag_a_row(const _Float16* base, int ld) {
  const int lane = threadIdx.x & 31;
  const int r  = lane & 15;
  const int kh = (lane >> 4) << 3;            // 0 or 8
  const _Float16* p0 = base + r * ld + kh;    // K = kh..kh+7
  const _Float16* p1 = p0 + 16;               // K = 16+kh..16+kh+7
  v16h f;
#pragma unroll
  for (int i = 0; i < 8; ++i) { f[i] = p0[i]; f[8 + i] = p1[i]; }
  return f;
}

// B-matrix 32x16 f16: lanes 0-15 hold K={0..7,16..23},
// lanes 16-31 hold K={8..15,24..31}, column N = lane&15 (matches A).
// B == Wrow^T where Wrow is row-major [N][K]; two contiguous 16B runs per lane.
DEVINL v16h frag_b_rowN(const _Float16* base, int ld) {
  const int lane = threadIdx.x & 31;
  const int n  = lane & 15;
  const int kb = (lane >> 4) << 3;            // 0 or 8
  const _Float16* p = base + n * ld + kb;
  v16h f;
#pragma unroll
  for (int i = 0; i < 8; ++i) { f[i] = p[i]; f[8 + i] = p[16 + i]; }
  return f;
}

DEVINL v8f wmma16(v16h a, v16h b, v8f c) {
  return __builtin_amdgcn_wmma_f32_16x16x32_f16(false, a, false, b, (short)0, c,
                                                false, false);
}

// C/D 16x16 f32 layout: VGPR i -> (M=i, N=lane) lanes 0-15; (M=8+i, N=lane-16).
DEVINL void store_tile_f32(float* base, int ld, v8f acc, float scale) {
  const int lane = threadIdx.x & 31;
  const int n  = lane & 15;
  const int m0 = (lane >> 4) << 3;
#pragma unroll
  for (int i = 0; i < 8; ++i) base[(size_t)(m0 + i) * ld + n] = acc[i] * scale;
}

// ---------------- kernel 1: 4x4 maxpool + channel LayerNorm ----------------
// grid (L, B, 2), block 256 (one thread per channel)
__global__ void pool_ln_kernel(const float* __restrict__ x1,
                               const float* __restrict__ x2,
                               const float* __restrict__ g,
                               const float* __restrict__ bt,
                               float* __restrict__ p) {
  const int sp = blockIdx.x;
  const int b  = blockIdx.y;
  const int st = blockIdx.z;
  const int c  = threadIdx.x;
  const float* x = st ? x2 : x1;
  const int hp = sp >> 5, wp = sp & 31;
  const float* xb = x + (((size_t)b * C + c) * H + hp * 4) * W + wp * 4;
  float m = -3.4e38f;
#pragma unroll
  for (int i = 0; i < 4; ++i)
#pragma unroll
    for (int j = 0; j < 4; ++j) m = fmaxf(m, xb[i * W + j]);
  __shared__ float r1[256], r2[256];
  r1[c] = m; r2[c] = m * m;
  __syncthreads();
  for (int o = 128; o > 0; o >>= 1) {
    if (c < o) { r1[c] += r1[c + o]; r2[c] += r2[c + o]; }
    __syncthreads();
  }
  const float mean = r1[0] * (1.0f / C);
  const float var  = r2[0] * (1.0f / C) - mean * mean;
  const float rs   = rsqrtf(var + EPS);
  p[(((size_t)st * B + b) * C + c) * L + sp] = g[c] * (m - mean) * rs + bt[c];
}

// ---------------- kernel 2: depthwise 3x3 conv + BN -> f16 -----------------
// q,k in row-major head layout [B,NH,L,D]; v written TRANSPOSED [B,NH,D,S]
// so the P x V WMMA B-fragments are contiguous. grid (C, B, 2), block 256
__global__ void dwconv_bn_kernel(const float* __restrict__ p,
    const float* __restrict__ qk_w, const float* __restrict__ qk_g,
    const float* __restrict__ qk_b, const float* __restrict__ qk_m,
    const float* __restrict__ qk_v,
    const float* __restrict__ v_w,  const float* __restrict__ v_g,
    const float* __restrict__ v_b,  const float* __restrict__ v_m,
    const float* __restrict__ v_v,
    _Float16* __restrict__ q, _Float16* __restrict__ k,
    _Float16* __restrict__ vt1, _Float16* __restrict__ vt2) {
  const int c = blockIdx.x, b = blockIdx.y, st = blockIdx.z;
  const float* pb = p + (((size_t)st * B + b) * C + c) * L;
  const float* wq = qk_w + c * 9;
  const float* wv = v_w  + c * 9;
  const float scq = qk_g[c] * rsqrtf(qk_v[c] + EPS);
  const float scv = v_g[c]  * rsqrtf(v_v[c]  + EPS);
  const float bq  = qk_b[c] - scq * qk_m[c];
  const float bv  = v_b[c]  - scv * v_m[c];
  _Float16* dq = (st == 0) ? q   : k;
  _Float16* dv = (st == 0) ? vt1 : vt2;
  const int nh = c >> 5, d = c & 31;
  for (int pix = threadIdx.x; pix < L; pix += blockDim.x) {
    const int hp = pix >> 5, wp = pix & 31;
    float aq = 0.f, av = 0.f;
#pragma unroll
    for (int dy = -1; dy <= 1; ++dy) {
      const int yy = hp + dy;
      if ((unsigned)yy >= (unsigned)HP) continue;
#pragma unroll
      for (int dx = -1; dx <= 1; ++dx) {
        const int xx = wp + dx;
        if ((unsigned)xx >= (unsigned)WP) continue;
        const float pv = pb[yy * WP + xx];
        const int wi = (dy + 1) * 3 + (dx + 1);
        aq += wq[wi] * pv;
        av += wv[wi] * pv;
      }
    }
    dq[(((size_t)b * NH + nh) * L + pix) * D + d] = (_Float16)(scq * aq + bq);
    dv[(((size_t)b * NH + nh) * D + d) * S + pix] = (_Float16)(scv * av + bv);
  }
}

// ---------------- kernel 3: weights f32 -> f16 ----------------
__global__ void cvt_w_kernel(const float* __restrict__ a, const float* __restrict__ b,
                             const float* __restrict__ c,
                             _Float16* __restrict__ ah, _Float16* __restrict__ bh,
                             _Float16* __restrict__ ch,
                             int na, int nb, int nc) {
  const int i = blockIdx.x * blockDim.x + threadIdx.x;
  if (i < na) ah[i] = (_Float16)a[i];
  if (i < nb) bh[i] = (_Float16)b[i];
  if (i < nc) ch[i] = (_Float16)c[i];
}

// ---------------- kernel 4: scores = t * Q K^T (per batch*head) -------------
// grid (64, 8, B*NH), block 256 = 8 waves; one WMMA per 16x16 tile (K=32)
__global__ void qk_wmma_kernel(const _Float16* __restrict__ Q,
                               const _Float16* __restrict__ Kh,
                               float* __restrict__ scores) {
  const int lt  = blockIdx.x;                    // l-tile
  const int wv  = threadIdx.x >> 5;
  const int stt = blockIdx.y * 8 + wv;           // s-tile
  const int bh  = blockIdx.z;
  const _Float16* qa = Q  + ((size_t)bh * L + lt  * 16) * D;
  const _Float16* kb = Kh + ((size_t)bh * L + stt * 16) * D;
  v16h a  = frag_a_row(qa, D);
  v16h bf = frag_b_rowN(kb, D);                  // B = K^T, K row-major [S][D]
  v8f acc = {};
  acc = wmma16(a, bf, acc);
  const float t = 0.17677669529663688f;          // 1/sqrt(32)
  store_tile_f32(scores + (size_t)bh * L * S + (size_t)(lt * 16) * S + stt * 16,
                 S, acc, t);
}

// ---------------- kernel 5: softmax over s -> P1 (f16) ----------------
// grid (L, B*NH), block 256; each thread owns 4 elements of its 1024-row
__global__ void softmax_s_kernel(const float* __restrict__ sc,
                                 _Float16* __restrict__ P1) {
  const size_t row = (size_t)blockIdx.y * L + blockIdx.x;
  const float* r = sc + row * S;
  const int t = threadIdx.x;
  float v[4];
#pragma unroll
  for (int j = 0; j < 4; ++j) v[j] = r[t + 256 * j];
  __shared__ float red[256];
  float m = fmaxf(fmaxf(v[0], v[1]), fmaxf(v[2], v[3]));
  red[t] = m;
  __syncthreads();
  for (int o = 128; o > 0; o >>= 1) {
    if (t < o) red[t] = fmaxf(red[t], red[t + o]);
    __syncthreads();
  }
  m = red[0];
  __syncthreads();
  float s = 0.f;
#pragma unroll
  for (int j = 0; j < 4; ++j) { v[j] = __expf(v[j] - m); s += v[j]; }
  red[t] = s;
  __syncthreads();
  for (int o = 128; o > 0; o >>= 1) {
    if (t < o) red[t] += red[t + o];
    __syncthreads();
  }
  const float inv = 1.0f / red[0];
  _Float16* ob = P1 + row * S;
#pragma unroll
  for (int j = 0; j < 4; ++j) ob[t + 256 * j] = (_Float16)(v[j] * inv);
}

// ---------------- kernel 6: softmax over heads -> P2 (f16) ----------------
// one thread per (b,l,s); 8 values strided by L*S; coalesced in s
__global__ void softmax_h_kernel(const float* __restrict__ sc,
                                 _Float16* __restrict__ P2) {
  const size_t idx = (size_t)blockIdx.x * blockDim.x + threadIdx.x; // < B*L*S
  const size_t ls = idx & ((size_t)L * S - 1);
  const size_t b  = idx >> 20;                 // L*S = 2^20
  const float* base = sc + b * (size_t)NH * L * S + ls;
  float v[NH];
  float m = -3.4e38f;
#pragma unroll
  for (int h = 0; h < NH; ++h) {
    v[h] = base[(size_t)h * L * S];
    m = fmaxf(m, v[h]);
  }
  float s = 0.f;
#pragma unroll
  for (int h = 0; h < NH; ++h) { v[h] = __expf(v[h] - m); s += v[h]; }
  const float inv = 1.f / s;
  _Float16* ob = P2 + b * (size_t)NH * L * S + ls;
#pragma unroll
  for (int h = 0; h < NH; ++h) ob[(size_t)h * L * S] = (_Float16)(v[h] * inv);
}

// ---------------- kernel 7: tiled transpose P2 [bh][l][s] -> [bh][s][l] -----
// 64x64 LDS tiles, coalesced both directions. grid (16, 16, 32), block 256
__global__ void transpose_p_kernel(const _Float16* __restrict__ Pin,
                                   _Float16* __restrict__ Pout) {
  __shared__ _Float16 tile[64][68];
  const int s0 = blockIdx.x * 64;
  const int l0 = blockIdx.y * 64;
  const int bh = blockIdx.z;
  const _Float16* src = Pin  + (size_t)bh * L * S;
  _Float16*       dst = Pout + (size_t)bh * L * S;
  for (int e = threadIdx.x; e < 64 * 64; e += 256) {
    const int r = e >> 6, cc = e & 63;
    tile[r][cc] = src[(size_t)(l0 + r) * S + s0 + cc];
  }
  __syncthreads();
  for (int e = threadIdx.x; e < 64 * 64; e += 256) {
    const int r = e >> 6, cc = e & 63;
    dst[(size_t)(s0 + r) * L + l0 + cc] = tile[cc][r];
  }
}

// ---------------- kernel 8: m = P x V^T-layout  (WMMA, K=1024) --------------
// A row-major [M][S], Vt row-major [D][S]; out f16 [B][L][C] (heads packed).
// grid (8, B*NH), block 256 = 8 waves; wave -> one 16-row strip, 2 N-tiles
__global__ void attn_pv_kernel(const _Float16* __restrict__ Pm,
                               const _Float16* __restrict__ Vt,
                               _Float16* __restrict__ Mh) {
  const int wv = threadIdx.x >> 5;
  const int mt = blockIdx.x * 8 + wv;            // 0..63
  const int bh = blockIdx.y;
  const int b = bh >> 3, nh = bh & 7;
  const _Float16* Pb = Pm + (size_t)bh * L * S + (size_t)(mt * 16) * S;
  const _Float16* Vb = Vt + (size_t)bh * D * S;
  v8f acc0 = {}, acc1 = {};
  for (int k0 = 0; k0 < S; k0 += 32) {
    v16h a  = frag_a_row(Pb + k0, S);
    v16h b0 = frag_b_rowN(Vb + k0, S);
    v16h b1 = frag_b_rowN(Vb + (size_t)16 * S + k0, S);
    acc0 = wmma16(a, b0, acc0);
    acc1 = wmma16(a, b1, acc1);
  }
  const int lane = threadIdx.x & 31;
  const int n  = lane & 15;
  const int m0 = (lane >> 4) << 3;
  _Float16* ob = Mh + ((size_t)(b * L + mt * 16)) * C + nh * D;
#pragma unroll
  for (int i = 0; i < 8; ++i) {
    ob[(size_t)(m0 + i) * C + n]      = (_Float16)acc0[i];
    ob[(size_t)(m0 + i) * C + 16 + n] = (_Float16)acc1[i];
  }
}

// ---------------- kernel 9: merge linear (WMMA) -> mm[st][b][c'][l] ---------
// A = m f16 [B*L][C], B = merge_w^T (merge_w row-major [C'][C]).
// grid (1024), block 256 = 8 waves; 8192 tiles total
__global__ void merge_wmma_kernel(const _Float16* __restrict__ M1h,
                                  const _Float16* __restrict__ M2h,
                                  const _Float16* __restrict__ mwh,
                                  float* __restrict__ mm) {
  const int wv = threadIdx.x >> 5;
  const int id = blockIdx.x * 8 + wv;            // 0..8191
  const int nt   = id & 15;
  const int rowt = (id >> 4) & 255;
  const int st   = id >> 12;
  const _Float16* mh = st ? M2h : M1h;
  const int row0 = rowt * 16;
  const int b = row0 >> 10, l0 = row0 & 1023;
  v8f acc = {};
  for (int k0 = 0; k0 < C; k0 += 32) {
    v16h a  = frag_a_row(mh + (size_t)row0 * C + k0, C);
    v16h bf = frag_b_rowN(mwh + (size_t)(nt * 16) * C + k0, C);
    acc = wmma16(a, bf, acc);
  }
  const int lane = threadIdx.x & 31;
  const int n  = lane & 15;
  const int m0 = (lane >> 4) << 3;
  // transposed store: mm plane-major for the upsample gather
  float* ob = mm + ((size_t)(st * B + b) * C + nt * 16 + n) * L + l0 + m0;
#pragma unroll
  for (int i = 0; i < 8; ++i) ob[i] = acc[i];
}

// ---------------- kernel 10: fused upsample + MLP (WMMA) + LN + residual ----
// grid (4096, 2), block 256 = 8 waves; 16 rows per block; bilinear x4 fused
// into the stage-0 gather (mm is L2-resident, no U tensor materialized).
__global__ void mlp_fuse_kernel(const float* __restrict__ x1,
                                const float* __restrict__ x2,
                                const float* __restrict__ mm,
                                const _Float16* __restrict__ w1h,
                                const _Float16* __restrict__ w2h,
                                const float* __restrict__ g2,
                                const float* __restrict__ b2,
                                float* __restrict__ out) {
  const int st = blockIdx.y;
  const float* x = st ? x2 : x1;
  float* ob = out + (size_t)st * B * C * H * W;
  const int row0 = blockIdx.x * 16;              // < B*H*W
  const int b    = row0 >> 14;                   // H*W = 16384
  const int rem  = row0 & 16383;
  const int y    = rem >> 7;
  const int xcol = rem & 127;                    // multiple of 16

  __shared__ _Float16 zt[16 * C2];               // input tile  (f16)
  __shared__ _Float16 ht[16 * C2];               // hidden tile (f16, post-ReLU)
  __shared__ float    ot[16 * C];                // output tile (f32)
  __shared__ float    redA[256], redB[256];
  __shared__ float    muS[16], rsS[16];

  // per-block vertical bilinear coefficients (y uniform across the 16 rows)
  const float sy = (y + 0.5f) * 0.25f - 0.5f;
  const int   yf = (int)floorf(sy);
  const float wy = sy - (float)yf;
  const int y0c = max(yf, 0);
  const int y1c = min(yf + 1, HP - 1);

  // stage 0: gather z = concat(x, upsample(mm)) for 16 rows, convert to f16
  for (int e = threadIdx.x; e < 16 * C2; e += 256) {
    const int c  = e >> 4;
    const int rr = e & 15;
    float val;
    if (c < C) {
      val = x[(((size_t)b * C + c) * H + y) * W + xcol + rr];
    } else {
      const float* src = mm + ((size_t)(st * B + b) * C + (c - C)) * L;
      const float sx = (float)(xcol + rr) * 0.25f + 0.125f - 0.5f;
      const int   xf = (int)floorf(sx);
      const float wx = sx - (float)xf;
      const int x0c = max(xf, 0);
      const int x1c = min(xf + 1, WP - 1);
      const float v00 = src[y0c * WP + x0c], v01 = src[y0c * WP + x1c];
      const float v10 = src[y1c * WP + x0c], v11 = src[y1c * WP + x1c];
      val = (1.f - wy) * ((1.f - wx) * v00 + wx * v01) +
            wy         * ((1.f - wx) * v10 + wx * v11);
    }
    zt[rr * C2 + c] = (_Float16)val;
  }
  __syncthreads();

  const int wv   = threadIdx.x >> 5;
  const int lane = threadIdx.x & 31;
  const int n    = lane & 15;
  const int m0   = (lane >> 4) << 3;

  // stage 1: hidden = relu(z @ w1^T); k-outer, 4 accumulators reuse one A-frag
  {
    v8f acc[4] = {};
    for (int k0 = 0; k0 < C2; k0 += 32) {
      v16h a = frag_a_row(&zt[k0], C2);
#pragma unroll
      for (int j = 0; j < 4; ++j) {
        v16h bf = frag_b_rowN(w1h + (size_t)((wv * 4 + j) * 16) * C2 + k0, C2);
        acc[j] = wmma16(a, bf, acc[j]);
      }
    }
#pragma unroll
    for (int j = 0; j < 4; ++j) {
      const int nt = wv * 4 + j;
#pragma unroll
      for (int i = 0; i < 8; ++i)
        ht[(m0 + i) * C2 + nt * 16 + n] = (_Float16)fmaxf(acc[j][i], 0.f);
    }
  }
  __syncthreads();

  // stage 2: o = hidden @ w2^T; k-outer, 2 accumulators reuse one A-frag
  {
    v8f acc[2] = {};
    for (int k0 = 0; k0 < C2; k0 += 32) {
      v16h a = frag_a_row(&ht[k0], C2);
#pragma unroll
      for (int j = 0; j < 2; ++j) {
        v16h bf = frag_b_rowN(w2h + (size_t)((wv * 2 + j) * 16) * C2 + k0, C2);
        acc[j] = wmma16(a, bf, acc[j]);
      }
    }
#pragma unroll
    for (int j = 0; j < 2; ++j) {
      const int nt = wv * 2 + j;
#pragma unroll
      for (int i = 0; i < 8; ++i)
        ot[(m0 + i) * C + nt * 16 + n] = acc[j][i];
    }
  }
  __syncthreads();

  // stage 3: LayerNorm over C per row + residual, store NCHW
  const int r   = threadIdx.x >> 4;              // row 0..15
  const int sub = threadIdx.x & 15;              // 16 threads per row
  float s = 0.f, s2 = 0.f;
#pragma unroll
  for (int kk = 0; kk < 16; ++kk) {
    const float vv = ot[r * C + sub * 16 + kk];
    s += vv; s2 += vv * vv;
  }
  redA[threadIdx.x] = s;
  redB[threadIdx.x] = s2;
  __syncthreads();
  if (sub == 0) {
    float ts = 0.f, ts2 = 0.f;
    for (int kk = 0; kk < 16; ++kk) { ts += redA[r * 16 + kk]; ts2 += redB[r * 16 + kk]; }
    const float mean = ts * (1.f / C);
    const float var  = ts2 * (1.f / C) - mean * mean;
    muS[r] = mean;
    rsS[r] = rsqrtf(var + EPS);
  }
  __syncthreads();
  const float mean = muS[r], rsv = rsS[r];
#pragma unroll
  for (int kk = 0; kk < 16; ++kk) {
    const int c = sub * 16 + kk;
    const float zn = g2[c] * (ot[r * C + c] - mean) * rsv + b2[c];
    const size_t oi = (((size_t)b * C + c) * H + y) * W + xcol + r;
    ob[oi] = x[oi] + zn;
  }
}

// ---------------- host launcher ----------------
extern "C" void kernel_launch(void* const* d_in, const int* in_sizes, int n_in,
                              void* d_out, int out_size, void* d_ws, size_t ws_size,
                              hipStream_t stream) {
  const float* x1     = (const float*)d_in[0];
  const float* x2     = (const float*)d_in[1];
  const float* qk_w   = (const float*)d_in[2];
  const float* qk_g   = (const float*)d_in[3];
  const float* qk_b   = (const float*)d_in[4];
  const float* qk_m   = (const float*)d_in[5];
  const float* qk_v   = (const float*)d_in[6];
  const float* v_w    = (const float*)d_in[7];
  const float* v_g    = (const float*)d_in[8];
  const float* v_b    = (const float*)d_in[9];
  const float* v_m    = (const float*)d_in[10];
  const float* v_v    = (const float*)d_in[11];
  const float* ln1_g  = (const float*)d_in[12];
  const float* ln1_b  = (const float*)d_in[13];
  const float* ln2_g  = (const float*)d_in[14];
  const float* ln2_b  = (const float*)d_in[15];
  const float* mergew = (const float*)d_in[16];
  const float* mlp_w1 = (const float*)d_in[17];
  const float* mlp_w2 = (const float*)d_in[18];
  float* out = (float*)d_out;

  // workspace carve-out
  char* ws = (char*)d_ws;
  auto carve = [&](size_t bytes) -> char* {
    char* r = ws;
    ws += (bytes + 255) & ~(size_t)255;
    return r;
  };
  float*    p_pool = (float*)   carve(2ull * B * C * L * sizeof(float));
  _Float16* Qh     = (_Float16*)carve((size_t)B * NH * L * D * sizeof(_Float16));
  _Float16* Kh     = (_Float16*)carve((size_t)B * NH * L * D * sizeof(_Float16));
  _Float16* Vt1    = (_Float16*)carve((size_t)B * NH * D * S * sizeof(_Float16));
  _Float16* Vt2    = (_Float16*)carve((size_t)B * NH * D * S * sizeof(_Float16));
  float*    scores = (float*)   carve((size_t)B * NH * L * S * sizeof(float));
  _Float16* P1     = (_Float16*)carve((size_t)B * NH * L * S * sizeof(_Float16));
  _Float16* P2     = (_Float16*)carve((size_t)B * NH * L * S * sizeof(_Float16));
  _Float16* P2t    = (_Float16*)carve((size_t)B * NH * L * S * sizeof(_Float16));
  _Float16* M1h    = (_Float16*)carve((size_t)B * L * C * sizeof(_Float16));
  _Float16* M2h    = (_Float16*)carve((size_t)B * L * C * sizeof(_Float16));
  float*    mm     = (float*)   carve(2ull * B * C * L * sizeof(float));
  _Float16* w1h    = (_Float16*)carve((size_t)C2 * C2 * sizeof(_Float16));
  _Float16* w2h    = (_Float16*)carve((size_t)C * C2 * sizeof(_Float16));
  _Float16* mwh    = (_Float16*)carve((size_t)C * C * sizeof(_Float16));

  // 1) pool + LN
  pool_ln_kernel<<<dim3(L, B, 2), 256, 0, stream>>>(x1, x2, ln1_g, ln1_b, p_pool);

  // 2) depthwise conv + BN -> Q,K (row) / V1,V2 (transposed), f16
  dwconv_bn_kernel<<<dim3(C, B, 2), 256, 0, stream>>>(
      p_pool, qk_w, qk_g, qk_b, qk_m, qk_v, v_w, v_g, v_b, v_m, v_v,
      Qh, Kh, Vt1, Vt2);

  // 3) weights to f16
  cvt_w_kernel<<<dim3((C2 * C2 + 255) / 256), 256, 0, stream>>>(
      mlp_w1, mlp_w2, mergew, w1h, w2h, mwh, C2 * C2, C * C2, C * C);

  // 4) scores = t * Q K^T  (WMMA)
  qk_wmma_kernel<<<dim3(L / 16, 8, B * NH), 256, 0, stream>>>(Qh, Kh, scores);

  // 5) softmax over s -> P1
  softmax_s_kernel<<<dim3(L, B * NH), 256, 0, stream>>>(scores, P1);

  // 6) softmax over heads -> P2, then tiled transpose -> P2t
  softmax_h_kernel<<<dim3((B * L * S) / 256), 256, 0, stream>>>(scores, P2);
  transpose_p_kernel<<<dim3(S / 64, L / 64, B * NH), 256, 0, stream>>>(P2, P2t);

  // 7) m1 = P1 x V2 ; m2 = P2^T x V1  (WMMA, all-contiguous fragments)
  attn_pv_kernel<<<dim3(8, B * NH), 256, 0, stream>>>(P1,  Vt2, M1h);
  attn_pv_kernel<<<dim3(8, B * NH), 256, 0, stream>>>(P2t, Vt1, M2h);

  // 8) merge linear (WMMA) -> mm (plane-major)
  merge_wmma_kernel<<<dim3(1024), 256, 0, stream>>>(M1h, M2h, mwh, mm);

  // 9) fused upsample + MLP + LN + residual (WMMA) -> out1, out2
  mlp_fuse_kernel<<<dim3((B * H * W) / 16, 2), 256, 0, stream>>>(
      x1, x2, mm, w1h, w2h, ln2_g, ln2_b, out);

  (void)in_sizes; (void)n_in; (void)out_size; (void)ws_size;
}
